// SelfAttention_11063835754610
// MI455X (gfx1250) — matrix operand order repaired and checked by reading, not compiled, not run on hardware
//
#include <hip/hip_runtime.h>
#include <hip/hip_bf16.h>

#define B_ 8
#define T_ 2048
#define E_ 1024

typedef __attribute__((ext_vector_type(16))) __bf16 v16bf;
typedef __attribute__((ext_vector_type(8)))  __bf16 v8bf;
typedef __attribute__((ext_vector_type(8)))  float  v8f;
typedef __attribute__((ext_vector_type(4)))  float  v4f;

// ---------------------------------------------------------------------------
// WMMA operand loaders (CDNA5 wave32 layouts, CDNA5 ISA §7.12.2)
//
// A (16x32 bf16, M x K): lane L -> m = L&15, half = L>>4.
//   element i holds k = 16*half + i (16 contiguous K, same order as B).
//   Shifting the lane base by 16*half makes the needed relative elements
//   exactly {0..15}: two fully-used 128-bit loads per lane.
// B (32x16 bf16, K x N): lane L -> n = L&15, half = L>>4.
//   element i holds k = 16*half + i (16 contiguous K): two 128-bit loads.
// C/D (16x16 f32): lane L -> n = L&15; VGPR r -> m = r + 8*(L>>4)
// ---------------------------------------------------------------------------

__device__ __forceinline__ v16bf load_A_bf16(const __bf16* __restrict__ src,
                                             int row0, int k0, int ld) {
  const int l = threadIdx.x & 31;
  const int m = l & 15, half = l >> 4;
  const __bf16* p = src + (size_t)(row0 + m) * ld + k0 + 16 * half;
  v8bf lo = *(const v8bf*)p;          // relative k 0..7
  v8bf hi = *(const v8bf*)(p + 8);    // relative k 8..15
  return __builtin_shufflevector(lo, hi, 0, 1, 2, 3, 4, 5, 6, 7,
                                 8, 9, 10, 11, 12, 13, 14, 15);
}

__device__ __forceinline__ v16bf load_B_bf16(const __bf16* __restrict__ p) {
  v8bf lo = *(const v8bf*)p;
  v8bf hi = *(const v8bf*)(p + 8);
  return __builtin_shufflevector(lo, hi, 0, 1, 2, 3, 4, 5, 6, 7,
                                 8, 9, 10, 11, 12, 13, 14, 15);
}

#define WMMA_BF16(A, Bm, C) \
  __builtin_amdgcn_wmma_f32_16x16x32_bf16(false, (A), false, (Bm), (short)0, (C), false, false)

// ---------------------------------------------------------------------------
// K1: QKV projections.  y[n,f] = sum_e X[n,e]*W[f,e] + b[f]
// One wave computes a 16x64 strip (4 N-tiles, A reused 4x per K-step).
// Inner loop is staged: issue ALL raw f32 b128 loads into distinct registers,
// then convert, then 4 WMMAs -> one load clause + one wait per K-step.
// Q,K stored row-major bf16; V stored transposed per batch: Vt[b][f][t].
// ---------------------------------------------------------------------------
__global__ void __launch_bounds__(256) qkv_gemm(
    const float* __restrict__ X,
    const float* __restrict__ Wk, const float* __restrict__ bk,
    const float* __restrict__ Wq, const float* __restrict__ bq,
    const float* __restrict__ Wv, const float* __restrict__ bv,
    __bf16* __restrict__ Qb, __bf16* __restrict__ Kb, __bf16* __restrict__ Vt) {
  const int tile = blockIdx.x * 8 + (threadIdx.x >> 5);  // [0, 16384)
  const int sup = tile & 15;   // feature supertile (64 features)
  const int rt  = tile >> 4;   // row tile (B*T/16 = 1024)
  const int row0 = rt * 16;
  const int f0 = sup * 64;

  const int sel = blockIdx.z;
  const float* W    = (sel == 0) ? Wk : (sel == 1) ? Wq : Wv;
  const float* bias = (sel == 0) ? bk : (sel == 1) ? bq : bv;

  const int l = threadIdx.x & 31;
  const int n = l & 15, half = l >> 4;

  const float* rowA = X + (size_t)(row0 + (l & 15)) * E_ + 16 * half;
  const float* rowB = W + (size_t)(f0 + n) * E_ + 16 * half;

  v8f cc[4] = {};
  for (int e0 = 0; e0 < E_; e0 += 32) {
    // ---- stage 1: issue all raw loads (distinct destinations) ----
    const float* pA = rowA + e0;
    v4f ra[4];
    ra[0] = *(const v4f*)pA;
    ra[1] = *(const v4f*)(pA + 4);
    ra[2] = *(const v4f*)(pA + 8);
    ra[3] = *(const v4f*)(pA + 12);
    v4f rb[4][4];
#pragma unroll
    for (int t = 0; t < 4; ++t) {
      const float* q = rowB + e0 + (size_t)t * 16 * E_;
      rb[t][0] = *(const v4f*)q;
      rb[t][1] = *(const v4f*)(q + 4);
      rb[t][2] = *(const v4f*)(q + 8);
      rb[t][3] = *(const v4f*)(q + 12);
    }
    __builtin_prefetch(rowB + e0 + 32, 0, 1);
    // ---- stage 2: convert to bf16 operands ----
    v16bf a;
#pragma unroll
    for (int i = 0; i < 4; ++i) {
      a[i]      = (__bf16)ra[0][i];
      a[4 + i]  = (__bf16)ra[1][i];
      a[8 + i]  = (__bf16)ra[2][i];
      a[12 + i] = (__bf16)ra[3][i];
    }
    v16bf bm[4];
#pragma unroll
    for (int t = 0; t < 4; ++t)
#pragma unroll
      for (int i = 0; i < 4; ++i) {
        bm[t][i]      = (__bf16)rb[t][0][i];
        bm[t][4 + i]  = (__bf16)rb[t][1][i];
        bm[t][8 + i]  = (__bf16)rb[t][2][i];
        bm[t][12 + i] = (__bf16)rb[t][3][i];
      }
    // ---- stage 3: matrix ops ----
#pragma unroll
    for (int t = 0; t < 4; ++t) cc[t] = WMMA_BF16(a, bm[t], cc[t]);
  }

  if (sel == 2) {
    const int bat = row0 / T_, t0 = row0 % T_;
#pragma unroll
    for (int t = 0; t < 4; ++t) {
      const float bb = bias[f0 + t * 16 + n];
      __bf16* D = Vt + (size_t)bat * E_ * T_ + (size_t)(f0 + t * 16 + n) * T_ + t0;
#pragma unroll
      for (int r = 0; r < 8; ++r) D[r + 8 * half] = (__bf16)(cc[t][r] + bb);
    }
  } else {
    __bf16* D = (sel == 0) ? Kb : Qb;
#pragma unroll
    for (int t = 0; t < 4; ++t) {
      const float bb = bias[f0 + t * 16 + n];
#pragma unroll
      for (int r = 0; r < 8; ++r)
        D[(size_t)(row0 + r + 8 * half) * E_ + f0 + t * 16 + n] =
            (__bf16)(cc[t][r] + bb);
    }
  }
}

// ---------------------------------------------------------------------------
// K2a: per-COLUMN (query-axis softmax!) running max & sum over causal i >= j,
// split into 8 query-range chunks for parallelism (8192 waves).
// ---------------------------------------------------------------------------
__global__ void __launch_bounds__(256) col_stats_partial(
    const __bf16* __restrict__ Qb, const __bf16* __restrict__ Kb,
    float* __restrict__ pm, float* __restrict__ ps) {
  const int lin = blockIdx.x * 8 + (threadIdx.x >> 5);  // [0, 8192)
  const int ch = lin & 7;
  const int jt = (lin >> 3) & 127;
  const int b  = lin >> 10;
  const int j0 = jt * 16;
  const int l = threadIdx.x & 31;
  const int n = l & 15, half = l >> 4;
  const int base = b * T_;
  const float NEG = -1e30f;

  const int it_begin = (jt > ch * 16) ? jt : ch * 16;
  const int it_end   = (ch + 1) * 16;

  float m_run = NEG, s_run = 0.f;

  for (int it = it_begin; it < it_end; ++it) {
    const int i0 = it * 16;
    v8f c = {};
    for (int e0 = 0; e0 < E_; e0 += 32) {
      v16bf a = load_A_bf16(Qb, base + i0, e0, E_);
      const __bf16* pB = Kb + (size_t)(base + j0 + n) * E_ + e0 + 16 * half;
      __builtin_prefetch(pB + 32, 0, 1);
      v16bf bm = load_B_bf16(pB);
      c = WMMA_BF16(a, bm, c);
    }
    float sv[8];
    float tmax = NEG;
#pragma unroll
    for (int r = 0; r < 8; ++r) {
      const int i = i0 + r + 8 * half;
      sv[r] = (i >= j0 + n) ? c[r] : NEG;
      tmax = fmaxf(tmax, sv[r]);
    }
    const float m_new = fmaxf(m_run, tmax);
    float acc = s_run * __expf(m_run - m_new);
#pragma unroll
    for (int r = 0; r < 8; ++r) {
      const int i = i0 + r + 8 * half;
      if (i >= j0 + n) acc += __expf(sv[r] - m_new);
    }
    m_run = m_new;
    s_run = acc;
  }
  const float m_o = __shfl_xor(m_run, 16, 32);
  const float s_o = __shfl_xor(s_run, 16, 32);
  const float m_f = fmaxf(m_run, m_o);
  const float s_f = s_run * __expf(m_run - m_f) + s_o * __expf(m_o - m_f);
  if (half == 0) {
    const int idx = ((b * 128 + jt) * 8 + ch) * 16 + n;
    pm[idx] = m_f;
    ps[idx] = s_f;
  }
}

// K2b: combine the 8 chunk partials per column.
__global__ void __launch_bounds__(256) col_stats_reduce(
    const float* __restrict__ pm, const float* __restrict__ ps,
    float* __restrict__ colmax, float* __restrict__ colsum) {
  const int idx = blockIdx.x * 256 + threadIdx.x;  // [0, B*T)
  if (idx >= B_ * T_) return;
  const int b = idx / T_, j = idx % T_;
  const int jt = j >> 4, n = j & 15;
  const int base = ((b * 128 + jt) * 8) * 16 + n;
  float m = -1e30f;
#pragma unroll
  for (int c = 0; c < 8; ++c) m = fmaxf(m, pm[base + c * 16]);
  float s = 0.f;
#pragma unroll
  for (int c = 0; c < 8; ++c) s += ps[base + c * 16] * __expf(pm[base + c * 16] - m);
  colmax[idx] = m;
  colsum[idx] = s;
}

// ---------------------------------------------------------------------------
// K3: P[i,j] = exp(S[i,j] - colmax[j]) / colsum[j] (0 where masked), bf16.
// One wave per 16x64 strip (4 j-tiles, A reused 4x). Upper-triangle strips
// zero-filled so K4 can use padded 32-wide K-steps.
// ---------------------------------------------------------------------------
__global__ void __launch_bounds__(256) p_compute(
    const __bf16* __restrict__ Qb, const __bf16* __restrict__ Kb,
    const float* __restrict__ colmax, const float* __restrict__ colsum,
    __bf16* __restrict__ P) {
  const int lin = blockIdx.x * 8 + (threadIdx.x >> 5);  // [0, 32768)
  const int js = lin & 31;           // j supertile (64 columns)
  const int it = (lin >> 5) & 127;
  const int b  = lin >> 12;
  const int l = threadIdx.x & 31;
  const int n = l & 15, half = l >> 4;
  const int i0 = it * 16, j0 = js * 64;
  __bf16* Pt = P + (size_t)b * T_ * T_;

  if (j0 > i0 + 15) {  // wave-uniform: whole strip above the diagonal
#pragma unroll
    for (int t = 0; t < 4; ++t)
#pragma unroll
      for (int r = 0; r < 8; ++r)
        Pt[(size_t)(i0 + r + 8 * half) * T_ + j0 + t * 16 + n] = (__bf16)0.f;
    return;
  }

  const int base = b * T_;
  v8f cc[4] = {};
  for (int e0 = 0; e0 < E_; e0 += 32) {
    v16bf a = load_A_bf16(Qb, base + i0, e0, E_);
    const __bf16* pB = Kb + (size_t)(base + j0 + n) * E_ + e0 + 16 * half;
    __builtin_prefetch(pB + 32, 0, 1);
    v16bf bm[4];
#pragma unroll
    for (int t = 0; t < 4; ++t) bm[t] = load_B_bf16(pB + (size_t)t * 16 * E_);
#pragma unroll
    for (int t = 0; t < 4; ++t) cc[t] = WMMA_BF16(a, bm[t], cc[t]);
  }
#pragma unroll
  for (int t = 0; t < 4; ++t) {
    const int jn = j0 + t * 16 + n;
    const float mj  = colmax[base + jn];
    const float inv = 1.0f / colsum[base + jn];
#pragma unroll
    for (int r = 0; r < 8; ++r) {
      const int i = i0 + r + 8 * half;
      const float p = (i >= jn) ? __expf(cc[t][r] - mj) * inv : 0.f;
      Pt[(size_t)i * T_ + jn] = (__bf16)p;
    }
  }
}

// ---------------------------------------------------------------------------
// K4: out[b,i,e] = sum_j P[i,j] * V[j,e]   (V pre-transposed: Vt[b][e][j])
// One wave per 16x64 output strip (4 e-tiles). Causal K-range only; odd tail
// covered by zero-filled P tiles.
// ---------------------------------------------------------------------------
__global__ void __launch_bounds__(256) out_gemm(
    const __bf16* __restrict__ P, const __bf16* __restrict__ Vt,
    float* __restrict__ out) {
  const int lin = blockIdx.x * 8 + (threadIdx.x >> 5);  // [0, 16384)
  const int es = lin & 15;           // e supertile (64 features)
  const int it = (lin >> 4) & 127;
  const int b  = lin >> 11;
  const int l = threadIdx.x & 31;
  const int n = l & 15, half = l >> 4;
  const int i0 = it * 16, e0 = es * 64;
  const __bf16* Pt = P + (size_t)b * T_ * T_;
  const __bf16* Vb = Vt + (size_t)b * E_ * T_;

  v8f cc[4] = {};
  const int nk = (it + 2) >> 1;  // 32-wide K steps over j-tiles 0..it(+pad)
  for (int kk = 0; kk < nk; ++kk) {
    const int j0 = kk * 32;
    v16bf a = load_A_bf16(Pt, i0, j0, T_);
    const __bf16* pB = Vb + (size_t)(e0 + n) * T_ + j0 + 16 * half;
    __builtin_prefetch(pB + 32, 0, 1);
    v16bf bm[4];
#pragma unroll
    for (int t = 0; t < 4; ++t) bm[t] = load_B_bf16(pB + (size_t)t * 16 * T_);
#pragma unroll
    for (int t = 0; t < 4; ++t) cc[t] = WMMA_BF16(a, bm[t], cc[t]);
  }
#pragma unroll
  for (int t = 0; t < 4; ++t)
#pragma unroll
    for (int r = 0; r < 8; ++r)
      out[(size_t)(b * T_ + i0 + r + 8 * half) * E_ + e0 + t * 16 + n] = cc[t][r];
}

// ---------------------------------------------------------------------------
extern "C" void kernel_launch(void* const* d_in, const int* in_sizes, int n_in,
                              void* d_out, int out_size, void* d_ws, size_t ws_size,
                              hipStream_t stream) {
  (void)in_sizes; (void)n_in; (void)out_size; (void)ws_size;
  const float* X  = (const float*)d_in[0];
  // d_in[1] (token ids) unused by the reference math
  const float* Wk = (const float*)d_in[2];
  const float* bk = (const float*)d_in[3];
  const float* Wq = (const float*)d_in[4];
  const float* bq = (const float*)d_in[5];
  const float* Wv = (const float*)d_in[6];
  const float* bv = (const float*)d_in[7];
  float* out = (float*)d_out;

  char* ws = (char*)d_ws;
  const size_t QK_BYTES = (size_t)B_ * T_ * E_ * 2;      // 32 MB each
  const size_t P_BYTES  = (size_t)B_ * T_ * T_ * 2;      // 64 MB
  __bf16* Qb = (__bf16*)(ws);
  __bf16* Kb = (__bf16*)(ws + QK_BYTES);
  __bf16* Vt = (__bf16*)(ws + 2 * QK_BYTES);
  __bf16* P  = (__bf16*)(ws + 3 * QK_BYTES);
  char* stats = ws + 3 * QK_BYTES + P_BYTES;
  float* colmax = (float*)(stats);                        // B*T
  float* colsum = colmax + B_ * T_;                       // B*T
  float* pm = colsum + B_ * T_;                           // 8*128*8*16
  float* ps = pm + B_ * 128 * 8 * 16;

  // K1: 16384 strips per matrix, 8 waves/block, z = {K,Q,V}
  qkv_gemm<<<dim3(2048, 1, 3), 256, 0, stream>>>(X, Wk, bk, Wq, bq, Wv, bv,
                                                 Qb, Kb, Vt);
  // K2a: 8 * 128 * 8 chunk-partials = 8192 waves
  col_stats_partial<<<dim3(1024), 256, 0, stream>>>(Qb, Kb, pm, ps);
  // K2b: one thread per column
  col_stats_reduce<<<dim3((B_ * T_ + 255) / 256), 256, 0, stream>>>(pm, ps,
                                                                    colmax, colsum);
  // K3: 8 * 128 * 32 strips = 32768 waves
  p_compute<<<dim3(4096), 256, 0, stream>>>(Qb, Kb, colmax, colsum, P);
  // K4: 8 * 128 * 16 strips = 16384 waves
  out_gemm<<<dim3(2048), 256, 0, stream>>>(P, Vt, out);
}
